// Attention_46986942218849
// MI455X (gfx1250) — compile-verified, matrix operand order checked by reading, not run on hardware
//
#include <hip/hip_runtime.h>
#include <math.h>

typedef __attribute__((ext_vector_type(16))) _Float16 v16h;
typedef __attribute__((ext_vector_type(8)))  _Float16 v8h;
typedef __attribute__((ext_vector_type(4)))  _Float16 v4h;
typedef __attribute__((ext_vector_type(8)))  float    v8f;

#define B_      2
#define S_      2048
#define H_      32
#define D_      128
#define HD_     (H_*D_)     // 4096
#define KVHD_   (8*D_)      // 1024
#define WIN_    1024
#define LOG2E   1.4426950408889634f

// padded LDS row strides (halves) to kill 64-bank stride conflicts
#define QK_LDW  136         // 128 + 8
#define VT_LDW  40          // 32 + 8
#define P_LDW   40          // 32 + 8

// 16-element f16 WMMA A/B fragment slice for this lane (CDNA5 16-bit layout):
// lanes 0-15 hold k-elems {0..7,16..23}, lanes 16-31 hold {8..15,24..31}.
__device__ __forceinline__ v16h frag16(const _Float16* rowp, int off) {
    union { v16h v; v8h h[2]; } u;
    u.h[0] = *(const v8h*)(rowp + off);
    u.h[1] = *(const v8h*)(rowp + off + 16);
    return u.v;
}

// One 32-key FA block. MW: apply window-edge mask; MC: apply causal-edge mask.
// Middle phase instantiates with <false,false> -> all mask code compiled out.
template<bool MW, bool MC>
__device__ __forceinline__ void fa_block(
    int kb, int tid, int ln, int hb, int off, int rowg0, int b, int kvh,
    const float* __restrict__ kg_base, const float* __restrict__ vg_base,
    _Float16* Ks, _Float16* Vts, _Float16* Pw,
    const v16h (&aq)[4], const v16h& ones,
    v8f (&o)[8], v8f& lsum, float (&mi)[8],
    float (&bias0)[8], float bshift16, float dbias)
{
    const v8f vzero = {};

    // ---- stage K tile (32x128, row-major f16), coalesced b128 reads
    const float4* kg = (const float4*)(kg_base + (size_t)(b * S_ + kb) * KVHD_ + kvh * D_);
    for (int it = tid; it < 32 * 32; it += 128) {
        int r_ = it >> 5, c_ = it & 31;
        float4 f = kg[(size_t)r_ * (KVHD_ / 4) + c_];
        v4h hv = { (_Float16)f.x, (_Float16)f.y, (_Float16)f.z, (_Float16)f.w };
        *(v4h*)(&Ks[r_ * QK_LDW + c_ * 4]) = hv;
    }
    // ---- stage V tile transposed (dim-major f16)
    const float4* vg = (const float4*)(vg_base + (size_t)(b * S_ + kb) * KVHD_ + kvh * D_);
    for (int it = tid; it < 32 * 32; it += 128) {
        int r_ = it >> 5, c_ = it & 31;       // key r_, dims 4c_..4c_+3
        float4 f = vg[(size_t)r_ * (KVHD_ / 4) + c_];
        int d0_ = c_ * 4;
        Vts[(d0_ + 0) * VT_LDW + r_] = (_Float16)f.x;
        Vts[(d0_ + 1) * VT_LDW + r_] = (_Float16)f.y;
        Vts[(d0_ + 2) * VT_LDW + r_] = (_Float16)f.z;
        Vts[(d0_ + 3) * VT_LDW + r_] = (_Float16)f.w;
    }
    __syncthreads();

    // ---- QK^T : two 16x16 score tiles, B-fragments software-pipelined so
    //      the d+1 loads are outstanding while wmma(d) issues (partial waits)
    v8f c0 = vzero, c1 = vzero;
    {
        const _Float16* krow0 = &Ks[ln * QK_LDW];
        const _Float16* krow1 = &Ks[(16 + ln) * QK_LDW];
        v16h bk0 = frag16(krow0, off);
        v16h bk1 = frag16(krow1, off);
        #pragma unroll
        for (int d = 0; d < 4; ++d) {
            v16h n0, n1;
            if (d < 3) {
                n0 = frag16(krow0 + (d + 1) * 32, off);
                n1 = frag16(krow1 + (d + 1) * 32, off);
            }
            c0 = __builtin_amdgcn_wmma_f32_16x16x32_f16(false, aq[d], false, bk0,
                                                        (short)0, c0, false, false);
            c1 = __builtin_amdgcn_wmma_f32_16x16x32_f16(false, aq[d], false, bk1,
                                                        (short)0, c1, false, false);
            bk0 = n0; bk1 = n1;
        }
    }

    // ---- bias (incremental ALiBi, log2 domain) + edge masks only
    float sc0[8], sc1[8];
    const int j0 = kb + ln, j1 = kb + 16 + ln;
    #pragma unroll
    for (int r = 0; r < 8; ++r) {
        float s0 = c0[r] + bias0[r];
        float s1 = c1[r] + (bias0[r] + bshift16);
        if (MW) {
            int gi = rowg0 + r;
            sc0[r] = (j0 >= gi - WIN_) ? s0 : -__builtin_inff();
            sc1[r] = (j1 >= gi - WIN_) ? s1 : -__builtin_inff();
        } else if (MC) {
            int gi = rowg0 + r;
            sc0[r] = (j0 <= gi) ? s0 : -__builtin_inff();
            sc1[r] = (j1 <= gi) ? s1 : -__builtin_inff();
        } else {
            sc0[r] = s0;
            sc1[r] = s1;
        }
        bias0[r] += dbias;
    }

    // ---- online softmax: row max over 32 keys (16-lane butterfly)
    float mnew[8], alpha[8];
    #pragma unroll
    for (int r = 0; r < 8; ++r) {
        float mr = fmaxf(sc0[r], sc1[r]);
        #pragma unroll
        for (int msk = 1; msk < 16; msk <<= 1)
            mr = fmaxf(mr, __shfl_xor(mr, msk, 32));
        mnew[r]  = fmaxf(mi[r], mr);
        alpha[r] = exp2f(mi[r] - mnew[r]);
        mi[r]    = mnew[r];
    }

    // ---- p = exp2(s - m); spill to LDS (C-layout -> A-layout transpose)
    #pragma unroll
    for (int r = 0; r < 8; ++r) {
        float p0 = exp2f(sc0[r] - mnew[r]);
        float p1 = exp2f(sc1[r] - mnew[r]);
        int row = hb * 8 + r;
        Pw[row * P_LDW + ln]      = (_Float16)p0;
        Pw[row * P_LDW + 16 + ln] = (_Float16)p1;
    }

    // ---- rescale O and lsum by alpha
    #pragma unroll
    for (int t = 0; t < 8; ++t)
        #pragma unroll
        for (int r = 0; r < 8; ++r) o[t][r] *= alpha[r];
    #pragma unroll
    for (int r = 0; r < 8; ++r) lsum[r] *= alpha[r];

    // ---- PV: O += P(16x32) x V(32x128); lsum += P x ones (row sums)
    //      V B-fragments double-buffered like the QK chain.
    {
        v16h ap = frag16(&Pw[ln * P_LDW], off);
        v16h bv = frag16(&Vts[ln * VT_LDW], off);
        #pragma unroll
        for (int t = 0; t < 8; ++t) {
            v16h nbv;
            if (t < 7) nbv = frag16(&Vts[((t + 1) * 16 + ln) * VT_LDW], off);
            o[t] = __builtin_amdgcn_wmma_f32_16x16x32_f16(false, ap, false, bv,
                                                          (short)0, o[t], false, false);
            bv = nbv;
        }
        lsum = __builtin_amdgcn_wmma_f32_16x16x32_f16(false, ap, false, ones,
                                                      (short)0, lsum, false, false);
    }
    __syncthreads();   // protect K/V tiles before next stage
}

__global__ __launch_bounds__(128)
void fa_swa_alibi_kernel(const float* __restrict__ q,
                         const float* __restrict__ k,
                         const float* __restrict__ v,
                         float* __restrict__ out) {
    __shared__ _Float16 Qs[64 * QK_LDW];
    __shared__ _Float16 Ks[32 * QK_LDW];
    __shared__ _Float16 Vts[128 * VT_LDW];      // V transposed: [dim][key]
    __shared__ _Float16 Ps[4 * 16 * P_LDW];     // per-wave P tile

    const int qt  = blockIdx.x;         // 0..31  (64-row q tile)
    const int h   = blockIdx.y;         // 0..31
    const int b   = blockIdx.z;         // 0..1
    const int q0  = qt * 64;
    const int kvh = h >> 2;             // GQA group of 4

    const int tid  = threadIdx.x;
    const int wave = tid >> 5;
    const int lane = tid & 31;
    const int ln   = lane & 15;
    const int hb   = lane >> 4;         // half-wave group (rows 0-7 vs 8-15)
    const int off  = hb ? 8 : 0;

    // fold softmax scale and log2(e) into Q; fold log2(e) into ALiBi slope
    const float qscale = (1.0f / 11.313708498984761f) * LOG2E;  // 1/sqrt(128)*log2e
    const float slope2 = exp2f(-0.25f * (float)(h + 1)) * LOG2E;

    // ---- stage Q tile (64x128 fp32 -> f16*scale) into LDS
    {
        const float4* qg = (const float4*)(q + (size_t)(b * S_ + q0) * HD_ + h * D_);
        for (int i = tid; i < 64 * 32; i += 128) {
            int r = i >> 5, c = i & 31;
            float4 f = qg[(size_t)r * (HD_ / 4) + c];
            v4h hv = { (_Float16)(f.x * qscale), (_Float16)(f.y * qscale),
                       (_Float16)(f.z * qscale), (_Float16)(f.w * qscale) };
            *(v4h*)(&Qs[r * QK_LDW + c * 4]) = hv;
        }
    }
    __syncthreads();

    // ---- resident Q A-fragments for this wave's 16 rows
    v16h aq[4];
    {
        const _Float16* qrow = &Qs[(wave * 16 + ln) * QK_LDW];
        #pragma unroll
        for (int d = 0; d < 4; ++d) aq[d] = frag16(qrow + d * 32, off);
    }
    v16h ones;
    #pragma unroll
    for (int e = 0; e < 16; ++e) ones[e] = (_Float16)1.0f;

    v8f o[8];
    const v8f vzero = {};
    #pragma unroll
    for (int t = 0; t < 8; ++t) o[t] = vzero;
    v8f lsum = vzero;
    float mi[8];
    #pragma unroll
    for (int r = 0; r < 8; ++r) mi[r] = -1e30f;

    const int rowg0 = q0 + wave * 16 + hb * 8;   // global seq row of slot 0

    // ---- phase bounds:
    //   [ks, kmid)   : window-edge masked (<= 2 iters)
    //   [kmid, q0)   : unmasked hot loop
    //   [q0, q0+64)  : causal-edge masked (2 iters)
    int ks   = q0 - WIN_;        if (ks < 0) ks = 0;
    int kmid = q0 - (WIN_ - 64); if (kmid < 0) kmid = 0;

    // incremental ALiBi bias registers (log2 domain); bias1 derived as
    // bias0 + slope2*16 to keep live registers down
    float bias0[8];
    const float dbias    = slope2 * 32.0f;
    const float bshift16 = slope2 * 16.0f;
    #pragma unroll
    for (int r = 0; r < 8; ++r) {
        int gi = rowg0 + r;
        bias0[r] = slope2 * (float)(ks + ln - gi);
    }

    _Float16* Pw = &Ps[wave * 16 * P_LDW];

    int kb = ks;
    for (; kb < kmid; kb += 32)
        fa_block<true, false>(kb, tid, ln, hb, off, rowg0, b, kvh, k, v,
                              Ks, Vts, Pw, aq, ones, o, lsum, mi,
                              bias0, bshift16, dbias);
    for (; kb < q0; kb += 32)
        fa_block<false, false>(kb, tid, ln, hb, off, rowg0, b, kvh, k, v,
                               Ks, Vts, Pw, aq, ones, o, lsum, mi,
                               bias0, bshift16, dbias);
    for (; kb < q0 + 64; kb += 32)
        fa_block<false, true>(kb, tid, ln, hb, off, rowg0, b, kvh, k, v,
                              Ks, Vts, Pw, aq, ones, o, lsum, mi,
                              bias0, bshift16, dbias);

    // ---- epilogue: O / l, fp32 store
    float* og = out + (size_t)(b * S_) * HD_ + h * D_;
    #pragma unroll
    for (int r = 0; r < 8; ++r) {
        float inv = 1.0f / lsum[r];
        int grow = rowg0 + r;
        #pragma unroll
        for (int t = 0; t < 8; ++t)
            og[(size_t)grow * HD_ + t * 16 + ln] = o[t][r] * inv;
    }
}

extern "C" void kernel_launch(void* const* d_in, const int* in_sizes, int n_in,
                              void* d_out, int out_size, void* d_ws, size_t ws_size,
                              hipStream_t stream) {
    const float* q = (const float*)d_in[0];
    const float* k = (const float*)d_in[1];
    const float* v = (const float*)d_in[2];
    float* out = (float*)d_out;
    dim3 grid(S_ / 64, H_, B_);
    fa_swa_alibi_kernel<<<grid, dim3(128), 0, stream>>>(q, k, v, out);
}
